// _RTGNActorRecurrent_39891656245842
// MI455X (gfx1250) — matrix operand order repaired and checked by reading, not compiled, not run on hardware
//
#include <hip/hip_runtime.h>

typedef unsigned short u16;
typedef unsigned int   u32;

typedef __attribute__((ext_vector_type(16))) __bf16 bf16x16;
typedef __attribute__((ext_vector_type(8)))  float  floatx8;

#define N_NODES 20000
#define N_EDGES 40000
#define HID     64
#define N_GRAPHS 500
#define NPG     40
#define N_TORS  5000   // N_GRAPHS * TORSIONS
#define ADIM    6

__device__ __forceinline__ u16 f2bf(float f) {
  u32 u = __float_as_uint(f);
  u32 r = u + 0x7FFFu + ((u >> 16) & 1u);   // round-to-nearest-even
  return (u16)(r >> 16);
}
__device__ __forceinline__ float bf2f(u16 u) {
  return __uint_as_float(((u32)u) << 16);
}
__device__ __forceinline__ float sigf(float x) { return 1.f / (1.f + __expf(-x)); }

// LDS byte address of a __shared__ object (flat LDS aperture keeps the
// wave-relative offset in the low 32 bits).
__device__ __forceinline__ u32 lds_addr(const void* p) {
  return (u32)(unsigned long long)p;
}
// CDNA5 async direct-to-LDS copy, 16B per lane, tracked by ASYNCcnt.
__device__ __forceinline__ void async_ld_b128(u32 lds, const void* gptr) {
  asm volatile("global_load_async_to_lds_b128 %0, %1, off"
               :: "v"(lds), "v"(gptr) : "memory");
}
__device__ __forceinline__ void wait_async0() {
  asm volatile("s_wait_asynccnt 0x0" ::: "memory");
}

// ---------------------------------------------------------------------------
// Generic bf16 WMMA GEMM:  C[M,N] = act( A[M,K] * B^T + bias[N] )
// A: bf16 row-major [M,K] (lda).  B: bf16 row-major [N,K] (ldb).
// Writes f32 (Cf) and/or bf16 (Cb), leading dim ldc.  relu optional.
// Block = 128 threads (4 wave32) -> 64x64 C tile; each wave owns a 16-row
// strip and 4 WMMA accumulators, K stepped by 32.  Full tiles are staged to
// LDS with global_load_async_to_lds_b128 (ASYNCcnt); ragged tiles fall back
// to guarded scalar loads.
// ---------------------------------------------------------------------------
__global__ __launch_bounds__(128) void k_gemm_bf16(
    const u16* __restrict__ A, int lda,
    const u16* __restrict__ B, int ldb,
    const float* __restrict__ bias,
    float* __restrict__ Cf, u16* __restrict__ Cb, int ldc,
    int M, int N, int K, int relu)
{
  __shared__ u16 As[64][40];   // 80B rows: 16B-aligned b128 targets, 4B-aligned
  __shared__ u16 Bs[64][40];   //           dword fragment reads (even k)

  const int tid  = threadIdx.x;
  const int lane = tid & 31;
  const int w    = tid >> 5;       // wave id 0..3 (wave32)
  const int half = lane >> 4;
  const int ml   = lane & 15;
  const int m0   = blockIdx.y * 64;
  const int n0   = blockIdx.x * 64;
  const int sr   = tid >> 1;       // staging row 0..63
  const int ss   = (tid & 1) * 16; // staging segment (16 u16)

  floatx8 acc[4] = {};

  for (int k0 = 0; k0 < K; k0 += 32) {
    const bool fullK = (k0 + 32 <= K);
    bool any_async = false;

    if (fullK && (m0 + 64 <= M)) {            // block-uniform fast path
      const u16* ga = A + (size_t)(m0 + sr) * lda + k0 + ss;
      u32 la = lds_addr(&As[sr][ss]);
      async_ld_b128(la, ga);
      async_ld_b128(la + 16u, ga + 8);
      any_async = true;
    } else {
      for (int idx = tid; idx < 64 * 32; idx += 128) {
        int mm = idx >> 5, kk = idx & 31;
        int gr = m0 + mm, gc = k0 + kk;
        As[mm][kk] = (gr < M && gc < K) ? A[(size_t)gr * lda + gc] : (u16)0;
      }
    }
    if (fullK && (n0 + 64 <= N)) {
      const u16* gb = B + (size_t)(n0 + sr) * ldb + k0 + ss;
      u32 lb = lds_addr(&Bs[sr][ss]);
      async_ld_b128(lb, gb);
      async_ld_b128(lb + 16u, gb + 8);
      any_async = true;
    } else {
      for (int idx = tid; idx < 64 * 32; idx += 128) {
        int nn = idx >> 5, kk = idx & 31;
        int gc = n0 + nn, gk = k0 + kk;
        Bs[nn][kk] = (gc < N && gk < K) ? B[(size_t)gc * ldb + gk] : (u16)0;
      }
    }
    if (any_async) wait_async0();
    __syncthreads();

    // A fragment (ISA 16-bit A 16x32 layout): lane m = w*16 + ml,
    // vgpr v holds K = 2v + (v>=4 ? 8:0) + half*8
    bf16x16 af;
    {
      const int mm = w * 16 + ml;
      #pragma unroll
      for (int v = 0; v < 8; v++) {
        int kb = 2 * v + (v >= 4 ? 8 : 0) + half * 8;
        u32 u = *(const u32*)&As[mm][kb];
        af[2 * v]     = __builtin_bit_cast(__bf16, (u16)(u & 0xFFFFu));
        af[2 * v + 1] = __builtin_bit_cast(__bf16, (u16)(u >> 16));
      }
    }
    #pragma unroll
    for (int nt = 0; nt < 4; nt++) {
      // B fragment (16-bit B KxN): lane n = nt*16+ml, same K order as A
      bf16x16 bfrag;
      const int nn = nt * 16 + ml;
      #pragma unroll
      for (int v = 0; v < 8; v++) {
        int kk = 2 * v + (v >= 4 ? 8 : 0) + half * 8;
        u32 u = *(const u32*)&Bs[nn][kk];
        bfrag[2 * v]     = __builtin_bit_cast(__bf16, (u16)(u & 0xFFFFu));
        bfrag[2 * v + 1] = __builtin_bit_cast(__bf16, (u16)(u >> 16));
      }
      acc[nt] = __builtin_amdgcn_wmma_f32_16x16x32_bf16(
          false, af, false, bfrag, (short)0, acc[nt], false, false);
    }
    __syncthreads();
  }

  // C/D layout: element r of v8f -> row = r + 8*half, col = lane%16
  #pragma unroll
  for (int nt = 0; nt < 4; nt++) {
    int col = n0 + nt * 16 + ml;
    #pragma unroll
    for (int r = 0; r < 8; r++) {
      int row = m0 + w * 16 + r + 8 * half;
      if (row < M && col < N) {
        float v = acc[nt][r];
        if (bias) v += bias[col];
        if (relu) v = fmaxf(v, 0.f);
        if (Cf) Cf[(size_t)row * ldc + col] = v;
        if (Cb) Cb[(size_t)row * ldc + col] = f2bf(v);
      }
    }
  }
}

// ---------------------------------------------------------------------------
// Small elementwise / scatter kernels
// ---------------------------------------------------------------------------
__global__ void k_fill(float* p, float v, int n) {
  int i = blockIdx.x * blockDim.x + threadIdx.x;
  if (i < n) p[i] = v;
}

__global__ void k_cast_bf16(const float* __restrict__ s, u16* __restrict__ d, int n) {
  int i = blockIdx.x * blockDim.x + threadIdx.x;
  if (i < n) d[i] = f2bf(s[i]);
}

// 64x64 transpose-cast: d[n*64+k] = bf16(s[k*64+n])   (conv_root -> [N,K])
__global__ void k_cast_tr64(const float* __restrict__ s, u16* __restrict__ d) {
  int i = blockIdx.x * blockDim.x + threadIdx.x;
  if (i >= 64 * 64) return;
  int n = i >> 6, k = i & 63;
  d[i] = f2bf(s[k * 64 + n]);
}

// out = relu(x @ lin0_w^T + b); h = out   (K=5, plain VALU)
__global__ void k_lin0(const float* __restrict__ x, const float* __restrict__ w,
                       const float* __restrict__ b, float* outf, u16* outb,
                       float* hf, u16* hb) {
  int i = blockIdx.x * blockDim.x + threadIdx.x;
  if (i >= N_NODES * HID) return;
  int n = i >> 6, h = i & 63;
  float a = b[h];
  #pragma unroll
  for (int d = 0; d < 5; d++) a += x[n * 5 + d] * w[h * 5 + d];
  a = fmaxf(a, 0.f);
  outf[i] = a; hf[i] = a;
  u16 v = f2bf(a);
  outb[i] = v; hb[i] = v;
}

// relu_e = bf16(relu(edge_attr @ enn_w1^T + b1))   (K=6, plain VALU)
__global__ void k_edge1(const float* __restrict__ ea, const float* __restrict__ w1,
                        const float* __restrict__ b1, u16* __restrict__ re) {
  int i = blockIdx.x * blockDim.x + threadIdx.x;
  if (i >= N_EDGES * 128) return;
  int e = i >> 7, j = i & 127;
  float a = b1[j];
  #pragma unroll
  for (int d = 0; d < 6; d++) a += ea[e * 6 + d] * w1[j * 6 + d];
  re[i] = f2bf(fmaxf(a, 0.f));
}

__global__ void k_degcount(const int* __restrict__ ei, float* deg) {
  int i = blockIdx.x * blockDim.x + threadIdx.x;
  if (i < N_EDGES) atomicAdd(&deg[ei[N_EDGES + i]], 1.f);
}
__global__ void k_deginv(float* deg) {
  int i = blockIdx.x * blockDim.x + threadIdx.x;
  if (i < N_NODES) deg[i] = 1.f / fmaxf(deg[i], 1.f);
}

// msg[e,k] = sum_h out[src,h]*ew[e,h,k]; scatter-add into agg[dst].
// 8 lanes per edge, 8 columns per lane, b128 row loads (128B/line per 8 lanes).
__global__ __launch_bounds__(256) void k_msg(
    const int* __restrict__ ei, const float* __restrict__ outf,
    const u16* __restrict__ ew, float* __restrict__ agg) {
  int tid = threadIdx.x;
  int e = blockIdx.x * 32 + (tid >> 3);
  int sub = tid & 7;
  if (e >= N_EDGES) return;
  int s = ei[e];
  int d = ei[N_EDGES + e];
  const u16* E = ew + (size_t)e * (HID * HID) + sub * 8;
  const float* X = outf + (size_t)s * HID;
  float acc[8] = {};
  for (int h = 0; h < HID; h++) {
    float xh = X[h];
    uint4 p = *(const uint4*)(E + (size_t)h * HID);
    acc[0] += xh * bf2f((u16)(p.x & 0xFFFFu));
    acc[1] += xh * bf2f((u16)(p.x >> 16));
    acc[2] += xh * bf2f((u16)(p.y & 0xFFFFu));
    acc[3] += xh * bf2f((u16)(p.y >> 16));
    acc[4] += xh * bf2f((u16)(p.z & 0xFFFFu));
    acc[5] += xh * bf2f((u16)(p.z >> 16));
    acc[6] += xh * bf2f((u16)(p.w & 0xFFFFu));
    acc[7] += xh * bf2f((u16)(p.w >> 16));
  }
  float* Ag = agg + (size_t)d * HID + sub * 8;
  #pragma unroll
  for (int j = 0; j < 8; j++) atomicAdd(&Ag[j], acc[j]);
}

// m = relu(agg/deg + root + conv_bias) -> bf16
__global__ void k_mrelu(const float* __restrict__ agg, const float* __restrict__ deginv,
                        const float* __restrict__ root, const float* __restrict__ cb,
                        u16* __restrict__ mb) {
  int i = blockIdx.x * blockDim.x + threadIdx.x;
  if (i >= N_NODES * HID) return;
  int n = i >> 6, h = i & 63;
  mb[i] = f2bf(fmaxf(agg[i] * deginv[n] + root[i] + cb[h], 0.f));
}

// GRU combine: h = (1-z)*n + z*h ; out = h
__global__ void k_gru(const float* __restrict__ gi, const float* __restrict__ gh,
                      float* hf, u16* hb, float* outf, u16* outb) {
  int i = blockIdx.x * blockDim.x + threadIdx.x;
  if (i >= N_NODES * HID) return;
  int n = i >> 6, h = i & 63;
  size_t b = (size_t)n * 192;
  float r  = sigf(gi[b + h] + gh[b + h]);
  float z  = sigf(gi[b + 64 + h] + gh[b + 64 + h]);
  float nn = tanhf(gi[b + 128 + h] + r * gh[b + 128 + h]);
  float hv = (1.f - z) * nn + z * hf[i];
  hf[i] = hv; outf[i] = hv;
  u16 v = f2bf(hv);
  hb[i] = v; outb[i] = v;
}

// LSTM gates: gates[g,j] = bih[j]+bhh[j] + q[g,:128].wih[j] + (hs? hs[g,:64].whh[j])
__global__ void k_lstm_gemm(const float* __restrict__ q, const float* __restrict__ hs,
                            const float* __restrict__ wih, const float* __restrict__ whh,
                            const float* __restrict__ bih, const float* __restrict__ bhh,
                            float* __restrict__ gates) {
  int i = blockIdx.x * blockDim.x + threadIdx.x;
  if (i >= N_GRAPHS * 256) return;
  int g = i >> 8, j = i & 255;
  float a = bih[j] + bhh[j];
  const float* qr = q + (size_t)g * 128;
  const float* wr = wih + (size_t)j * 128;
  #pragma unroll 4
  for (int k = 0; k < 128; k++) a += qr[k] * wr[k];
  if (hs) {
    const float* hr = hs + (size_t)g * 64;
    const float* vr = whh + (size_t)j * 64;
    #pragma unroll 4
    for (int k = 0; k < 64; k++) a += hr[k] * vr[k];
  }
  gates[i] = a;
}

// LSTM pointwise: c = sig(f)*c + sig(i)*tanh(g); h = sig(o)*tanh(c)
__global__ void k_lstm_pw(const float* __restrict__ gates, const float* __restrict__ cin,
                          float* __restrict__ hout, float* __restrict__ cout) {
  int i = blockIdx.x * blockDim.x + threadIdx.x;
  if (i >= N_GRAPHS * HID) return;
  int g = i >> 6, h = i & 63;
  size_t b = (size_t)g * 256;
  float ii = sigf(gates[b + h]);
  float ff = sigf(gates[b + 64 + h]);
  float gg = tanhf(gates[b + 128 + h]);
  float oo = sigf(gates[b + 192 + h]);
  float c = cin ? cin[i] : 0.f;
  c = ff * c + ii * gg;
  cout[i] = c;
  hout[i] = oo * tanhf(c);
}

// Set2Set attention: per-graph softmax over its 40 nodes, q_star = [hs, r]
__global__ __launch_bounds__(64) void k_attn(const float* __restrict__ outf,
                                             const float* __restrict__ hs,
                                             float* __restrict__ q_star) {
  int g = blockIdx.x;
  int t = threadIdx.x;
  __shared__ float sh[64];
  __shared__ float red[2];
  float e = -1e30f;
  if (t < NPG) {
    float a = 0.f;
    const float* o = outf + ((size_t)g * NPG + t) * HID;
    const float* hv = hs + (size_t)g * HID;
    #pragma unroll 4
    for (int h = 0; h < HID; h++) a += o[h] * hv[h];
    e = a;
  }
  sh[t] = e;
  __syncthreads();
  if (t == 0) {
    float mx = -1e30f;
    for (int i = 0; i < NPG; i++) mx = fmaxf(mx, sh[i]);
    red[0] = mx;
  }
  __syncthreads();
  float a = (t < NPG) ? __expf(sh[t] - red[0]) : 0.f;
  __syncthreads();
  sh[t] = a;
  __syncthreads();
  if (t == 0) {
    float s = 0.f;
    for (int i = 0; i < NPG; i++) s += sh[i];
    red[1] = s;
  }
  __syncthreads();
  float inv = 1.f / red[1];
  float r = 0.f;
  for (int i = 0; i < NPG; i++) r += sh[i] * outf[((size_t)g * NPG + i) * HID + t];
  q_star[(size_t)g * 128 + 64 + t] = r * inv;
  q_star[(size_t)g * 128 + t] = hs[(size_t)g * HID + t];
}

// feat[t, 0:64] = hx[t/10]; feat[t, 64+64q+h] = out[nonring[t,q], h]  (bf16)
__global__ void k_feat(const float* __restrict__ hx, const float* __restrict__ outf,
                       const int* __restrict__ nonring, u16* __restrict__ feat) {
  int i = blockIdx.x * blockDim.x + threadIdx.x;
  if (i >= N_TORS * 320) return;
  int t = i / 320, c = i % 320;
  float v;
  if (c < 64) {
    v = hx[(size_t)(t / 10) * HID + c];
  } else {
    int q = (c - 64) >> 6, h = (c - 64) & 63;
    int idx = nonring[t * 4 + q];
    v = outf[(size_t)idx * HID + h];
  }
  feat[i] = f2bf(v);
}

// logits = act @ mlp_w2^T + b2  (N=6, VALU)
__global__ void k_mlp2(const u16* __restrict__ act, const float* __restrict__ w2,
                       const float* __restrict__ b2, float* __restrict__ out) {
  int i = blockIdx.x * blockDim.x + threadIdx.x;
  if (i >= N_TORS * ADIM) return;
  int t = i / ADIM, a = i % ADIM;
  float s = b2[a];
  #pragma unroll 4
  for (int h = 0; h < HID; h++) s += bf2f(act[(size_t)t * HID + h]) * w2[a * HID + h];
  out[i] = s;
}

// ---------------------------------------------------------------------------
extern "C" void kernel_launch(void* const* d_in, const int* in_sizes, int n_in,
                              void* d_out, int out_size, void* d_ws, size_t ws_size,
                              hipStream_t stream) {
  const float* x         = (const float*)d_in[0];
  const float* edge_attr = (const float*)d_in[1];
  const int*   edge_idx  = (const int*)d_in[2];
  // d_in[3] batch, d_in[4] nrbidx: contiguous layouts -> recomputed in-kernel
  const int*   nonring   = (const int*)d_in[5];
  const float* lin0_w = (const float*)d_in[6];  const float* lin0_b = (const float*)d_in[7];
  const float* enn_w1 = (const float*)d_in[8];  const float* enn_b1 = (const float*)d_in[9];
  const float* enn_w2 = (const float*)d_in[10]; const float* enn_b2 = (const float*)d_in[11];
  const float* conv_root = (const float*)d_in[12]; const float* conv_bias = (const float*)d_in[13];
  const float* gru_wih = (const float*)d_in[14]; const float* gru_whh = (const float*)d_in[15];
  const float* gru_bih = (const float*)d_in[16]; const float* gru_bhh = (const float*)d_in[17];
  const float* s2s_wih = (const float*)d_in[18]; const float* s2s_whh = (const float*)d_in[19];
  const float* s2s_bih = (const float*)d_in[20]; const float* s2s_bhh = (const float*)d_in[21];
  const float* mem_wih = (const float*)d_in[22]; /* mem_whh unused: h0 == 0 */
  const float* mem_bih = (const float*)d_in[24]; const float* mem_bhh = (const float*)d_in[25];
  const float* mlp_w1 = (const float*)d_in[26]; const float* mlp_b1 = (const float*)d_in[27];
  const float* mlp_w2 = (const float*)d_in[28]; const float* mlp_b2 = (const float*)d_in[29];

  float* dout = (float*)d_out;
  float* hx = dout + N_GRAPHS * 10 * ADIM;          // 30000
  float* cx = hx + N_GRAPHS * HID;                  // 62000

  // --- carve workspace ---
  size_t off = 0;
  char* base = (char*)d_ws;
  auto carve = [&](size_t bytes) -> void* {
    void* p = base + off;
    off = (off + bytes + 255) & ~(size_t)255;
    return p;
  };
  u16*   ew      = (u16*)carve((size_t)N_EDGES * HID * HID * 2);   // 328 MB (bf16)
  u16*   relu_e  = (u16*)carve((size_t)N_EDGES * 128 * 2);
  float* outf    = (float*)carve((size_t)N_NODES * HID * 4);
  u16*   outb    = (u16*)carve((size_t)N_NODES * HID * 2);
  float* hf      = (float*)carve((size_t)N_NODES * HID * 4);
  u16*   hb      = (u16*)carve((size_t)N_NODES * HID * 2);
  float* agg     = (float*)carve((size_t)N_NODES * HID * 4);
  float* root    = (float*)carve((size_t)N_NODES * HID * 4);
  u16*   mb      = (u16*)carve((size_t)N_NODES * HID * 2);
  float* gi      = (float*)carve((size_t)N_NODES * 192 * 4);
  float* gh      = (float*)carve((size_t)N_NODES * 192 * 4);
  float* deginv  = (float*)carve((size_t)N_NODES * 4);
  u16*   w2b     = (u16*)carve((size_t)4096 * 128 * 2);
  u16*   rootTb  = (u16*)carve((size_t)64 * 64 * 2);
  u16*   gwihb   = (u16*)carve((size_t)192 * 64 * 2);
  u16*   gwhhb   = (u16*)carve((size_t)192 * 64 * 2);
  u16*   mlp1b   = (u16*)carve((size_t)64 * 320 * 2);
  float* q_star  = (float*)carve((size_t)N_GRAPHS * 128 * 4);
  float* hs      = (float*)carve((size_t)N_GRAPHS * HID * 4);
  float* cs      = (float*)carve((size_t)N_GRAPHS * HID * 4);
  float* gates   = (float*)carve((size_t)N_GRAPHS * 256 * 4);
  u16*   featb   = (u16*)carve((size_t)N_TORS * 320 * 2);
  u16*   actb    = (u16*)carve((size_t)N_TORS * HID * 2);

  auto blk = [](int n) { return (n + 255) / 256; };

  // --- weight casts to bf16 ([N,K] row-major for the GEMM) ---
  k_cast_bf16<<<blk(4096 * 128), 256, 0, stream>>>(enn_w2, w2b, 4096 * 128);
  k_cast_tr64<<<blk(64 * 64), 256, 0, stream>>>(conv_root, rootTb);
  k_cast_bf16<<<blk(192 * 64), 256, 0, stream>>>(gru_wih, gwihb, 192 * 64);
  k_cast_bf16<<<blk(192 * 64), 256, 0, stream>>>(gru_whh, gwhhb, 192 * 64);
  k_cast_bf16<<<blk(64 * 320), 256, 0, stream>>>(mlp_w1, mlp1b, 64 * 320);

  // --- lin0 + edge MLP first layer ---
  k_lin0<<<blk(N_NODES * HID), 256, 0, stream>>>(x, lin0_w, lin0_b, outf, outb, hf, hb);
  k_edge1<<<blk(N_EDGES * 128), 256, 0, stream>>>(edge_attr, enn_w1, enn_b1, relu_e);

  // --- edge network big GEMM (WMMA): ew = relu_e @ enn_w2^T + b2  (bf16 out) ---
  {
    dim3 g(4096 / 64, N_EDGES / 64);
    k_gemm_bf16<<<g, 128, 0, stream>>>(relu_e, 128, w2b, 128,
                                       enn_b2, nullptr, ew, 4096,
                                       N_EDGES, 4096, 128, 0);
  }

  // --- degree ---
  k_fill<<<blk(N_NODES), 256, 0, stream>>>(deginv, 0.f, N_NODES);
  k_degcount<<<blk(N_EDGES), 256, 0, stream>>>(edge_idx, deginv);
  k_deginv<<<blk(N_NODES), 256, 0, stream>>>(deginv);

  // --- 6 rounds of NNConv(mean) + GRU ---
  for (int it = 0; it < 6; it++) {
    k_fill<<<blk(N_NODES * HID), 256, 0, stream>>>(agg, 0.f, N_NODES * HID);
    k_msg<<<N_EDGES / 32, 256, 0, stream>>>(edge_idx, outf, ew, agg);
    {
      dim3 g(1, (N_NODES + 63) / 64);   // root = out @ conv_root  ([N,K]=rootT)
      k_gemm_bf16<<<g, 128, 0, stream>>>(outb, 64, rootTb, 64,
                                         nullptr, root, nullptr, 64,
                                         N_NODES, 64, 64, 0);
    }
    k_mrelu<<<blk(N_NODES * HID), 256, 0, stream>>>(agg, deginv, root, conv_bias, mb);
    {
      dim3 g(3, (N_NODES + 63) / 64);   // gi = m @ gru_wih^T + bih
      k_gemm_bf16<<<g, 128, 0, stream>>>(mb, 64, gwihb, 64,
                                         gru_bih, gi, nullptr, 192,
                                         N_NODES, 192, 64, 0);
      // gh = h @ gru_whh^T + bhh
      k_gemm_bf16<<<g, 128, 0, stream>>>(hb, 64, gwhhb, 64,
                                         gru_bhh, gh, nullptr, 192,
                                         N_NODES, 192, 64, 0);
    }
    k_gru<<<blk(N_NODES * HID), 256, 0, stream>>>(gi, gh, hf, hb, outf, outb);
  }

  // --- Set2Set (6 steps) ---
  k_fill<<<blk(N_GRAPHS * 128), 256, 0, stream>>>(q_star, 0.f, N_GRAPHS * 128);
  k_fill<<<blk(N_GRAPHS * HID), 256, 0, stream>>>(hs, 0.f, N_GRAPHS * HID);
  k_fill<<<blk(N_GRAPHS * HID), 256, 0, stream>>>(cs, 0.f, N_GRAPHS * HID);
  for (int it = 0; it < 6; it++) {
    k_lstm_gemm<<<blk(N_GRAPHS * 256), 256, 0, stream>>>(q_star, hs, s2s_wih, s2s_whh,
                                                         s2s_bih, s2s_bhh, gates);
    k_lstm_pw<<<blk(N_GRAPHS * HID), 256, 0, stream>>>(gates, cs, hs, cs);
    k_attn<<<N_GRAPHS, 64, 0, stream>>>(outf, hs, q_star);
  }

  // --- memory LSTM (one step, zero state) ---
  k_lstm_gemm<<<blk(N_GRAPHS * 256), 256, 0, stream>>>(q_star, nullptr, mem_wih, nullptr,
                                                       mem_bih, mem_bhh, gates);
  k_lstm_pw<<<blk(N_GRAPHS * HID), 256, 0, stream>>>(gates, nullptr, hx, cx);

  // --- gather features + MLP head ---
  k_feat<<<blk(N_TORS * 320), 256, 0, stream>>>(hx, outf, nonring, featb);
  {
    dim3 g(1, (N_TORS + 63) / 64);   // act = relu(feat @ mlp_w1^T + b1)
    k_gemm_bf16<<<g, 128, 0, stream>>>(featb, 320, mlp1b, 320,
                                       mlp_b1, nullptr, actb, 64,
                                       N_TORS, 64, 320, 1);
  }
  k_mlp2<<<blk(N_TORS * ADIM), 256, 0, stream>>>(actb, mlp_w2, mlp_b2, dout);
}
